// CorrectionNetKapSet_16226386444586
// MI455X (gfx1250) — matrix operand order repaired and checked by reading, not compiled, not run on hardware
//
#include <hip/hip_runtime.h>

#define DEVINL __device__ __forceinline__

typedef __attribute__((ext_vector_type(16))) __bf16 bf16x16;
typedef __attribute__((ext_vector_type(8)))  float  f32x8;

union AFrag {
  bf16x16 v;
  uint4   q[2];
};

constexpr int NN = 32768;
constexpr int EE = 1048576;
constexpr int HH = 64;
constexpr int TT = 5;

// ---- packed-weight region layout (dword offsets) ----
// fragment layout per matrix: [(kt*4+nt)*32 + lane] * 8 dwords
constexpr size_t PW_NE2 = 0;       // Wne2  64x64   : 2 ktiles -> 2048 dw
constexpr size_t PW_EE2 = 2048;    // Wee2  64x64
constexpr size_t PW_EU1 = 4096;    // Weu1 192x64   : 6 ktiles -> 6144 dw
constexpr size_t PW_EU2 = 10240;   // Weu2  64x64
constexpr size_t PW_NU1 = 12288;   // Wnu1 128x64   : 4 ktiles -> 4096 dw
constexpr size_t PW_NU2 = 16384;   // Wnu2  64x64
constexpr size_t PW_D1  = 18432;   // Wd1   64x64
constexpr size_t PW_TOTAL = 20480;

// ---- workspace byte layout ----
constexpr size_t OFF_NORM = 0;                              // 1 dword (abs-max bits)
constexpr size_t OFF_PW   = 256;                            // packed weights (80 KB)
constexpr size_t OFF_N    = OFF_PW + PW_TOTAL * 4;          // node feats bf16 [N,64]  (4 MB)
constexpr size_t OFF_AGG  = OFF_N + (size_t)NN * HH * 2;    // agg f32 [N,64]          (8 MB)
constexpr size_t OFF_E    = OFF_AGG + (size_t)NN * HH * 4;  // edge feats bf16 [E,64]  (128 MB)

DEVINL f32x8 wmma_bf16(bf16x16 a, bf16x16 b, f32x8 c) {
  return __builtin_amdgcn_wmma_f32_16x16x32_bf16(false, a, false, b, (short)0, c,
                                                 false, false);
}

DEVINL unsigned pack2(float lo, float hi) {
  union { __bf16 b; unsigned short u; } cl, ch;
  cl.b = (__bf16)lo; ch.b = (__bf16)hi;
  return (unsigned)cl.u | ((unsigned)ch.u << 16);
}

// A-fragment (16x32 bf16) load from a row-major 64-wide bf16 row.
// ISA layout: VGPR0-3 = K = kbase+8*half .. +7 ; VGPR4-7 = K = kbase+16+8*half .. +7
DEVINL bf16x16 load_afrag(const __bf16* row, int kbase, int half) {
  AFrag f;
  f.q[0] = *(const uint4*)(row + kbase + 8 * half);
  f.q[1] = *(const uint4*)(row + kbase + 16 + 8 * half);
  return f.v;
}

// B-fragment (32x16 bf16) load from pre-packed weights.
DEVINL bf16x16 load_bfrag(const unsigned* pw, int kt, int nt, int lane) {
  const uint4* p = (const uint4*)(pw + ((size_t)(kt * 4 + nt) * 32 + lane) * 8);
  AFrag f;
  f.q[0] = p[0];
  f.q[1] = p[1];
  return f.v;
}

// ---------------- init / reduction kernels ----------------

__global__ void k_zero(float* __restrict__ agg, unsigned* __restrict__ norm_bits) {
  size_t i = (size_t)blockIdx.x * blockDim.x + threadIdx.x;
  if (i < (size_t)NN * HH) agg[i] = 0.0f;
  if (i == 0) *norm_bits = 0u;
}

__global__ void k_absmax(const float* __restrict__ x, unsigned* __restrict__ out) {
  __shared__ unsigned red[256];
  unsigned m = 0u;
  for (int i = blockIdx.x * blockDim.x + threadIdx.x; i < EE;
       i += gridDim.x * blockDim.x) {
    unsigned b = __float_as_uint(fabsf(x[i]));
    m = (b > m) ? b : m;
  }
  red[threadIdx.x] = m;
  __syncthreads();
  for (int s = 128; s > 0; s >>= 1) {
    if ((int)threadIdx.x < s) {
      unsigned o = red[threadIdx.x + s];
      if (o > red[threadIdx.x]) red[threadIdx.x] = o;
    }
    __syncthreads();
  }
  if (threadIdx.x == 0) atomicMax(out, red[0]);
}

// Pack a [K,64] f32 weight matrix into WMMA B-fragments (bf16).
// grid = ktiles, block = 128 (4 ntiles x 32 lanes)
__global__ void k_pack(const float* __restrict__ W, unsigned* __restrict__ dst) {
  int kt   = blockIdx.x;
  int nt   = (threadIdx.x >> 5) & 3;
  int lane = threadIdx.x & 31;
  int col  = nt * 16 + (lane & 15);
  int kb   = kt * 32 + 8 * (lane >> 4);
  unsigned* o = dst + ((size_t)(kt * 4 + nt) * 32 + lane) * 8;
#pragma unroll
  for (int v = 0; v < 8; ++v) {
    int k0 = kb + 2 * v + (v >= 4 ? 8 : 0);
    o[v] = pack2(W[(size_t)k0 * HH + col], W[(size_t)(k0 + 1) * HH + col]);
  }
}

// ---------------- encoder (shared by node & edge) ----------------
// layer1 (1->H) is a per-lane broadcast FMA computed straight into A-fragments;
// layer2 (H->H) is WMMA.  One wave = 16 rows.
__global__ void k_encode(const float* __restrict__ x_in,
                         const float* __restrict__ W1, const float* __restrict__ b1,
                         const unsigned* __restrict__ pw2, const float* __restrict__ b2,
                         __bf16* __restrict__ out,
                         const unsigned* __restrict__ norm_bits) {
  int lane = threadIdx.x & 31;
  int w    = threadIdx.x >> 5;
  int tile = blockIdx.x * 8 + w;
  int half = lane >> 4;
  int lrow = lane & 15;
  int row  = tile * 16 + lrow;
  float x = x_in[row];
  if (norm_bits) x /= __uint_as_float(*norm_bits);

  f32x8 acc[4] = {};
#pragma unroll
  for (int kt = 0; kt < 2; ++kt) {
    bf16x16 a;
#pragma unroll
    for (int j = 0; j < 8; ++j) {
      int k0 = kt * 32 + 8 * half + j;
      int k1 = k0 + 16;
      float h0 = x * W1[k0] + b1[k0]; h0 = h0 > 0.f ? h0 : 0.f;
      float h1 = x * W1[k1] + b1[k1]; h1 = h1 > 0.f ? h1 : 0.f;
      a[j]     = (__bf16)h0;
      a[8 + j] = (__bf16)h1;
    }
#pragma unroll
    for (int nt = 0; nt < 4; ++nt)
      acc[nt] = wmma_bf16(a, load_bfrag(pw2, kt, nt, lane), acc[nt]);
  }
#pragma unroll
  for (int nt = 0; nt < 4; ++nt) {
    int col = nt * 16 + lrow;
    float bias = b2[col];
#pragma unroll
    for (int r = 0; r < 8; ++r) {
      int orow = tile * 16 + r + 8 * half;
      out[(size_t)orow * HH + col] = (__bf16)(acc[nt][r] + bias);
    }
  }
}

// ---------------- edge update (dominant kernel) ----------------
// One wave = 16 edges.  A = [e | n[s] | n[r]] (K=192), layer1 = 24 WMMA,
// LDS transpose, layer2 = 8 WMMA, then bf16 store + f32 atomic scatter (segment_sum).
__global__ void k_edge_update(__bf16* __restrict__ e, const __bf16* __restrict__ n,
                              const int* __restrict__ senders,
                              const int* __restrict__ receivers,
                              const unsigned* __restrict__ pw1, const float* __restrict__ b1,
                              const unsigned* __restrict__ pw2, const float* __restrict__ b2,
                              float* __restrict__ agg) {
  __shared__ __align__(16) __bf16 s_h[8][16 * HH];
  int lane = threadIdx.x & 31;
  int w    = threadIdx.x >> 5;
  int tile = blockIdx.x * 8 + w;
  int half = lane >> 4;
  int lrow = lane & 15;
  int edge = tile * 16 + lrow;
  int sid = senders[edge];
  int rid = receivers[edge];
  const __bf16* rowE = e + (size_t)edge * HH;
  const __bf16* rowS = n + (size_t)sid * HH;
  const __bf16* rowR = n + (size_t)rid * HH;

  f32x8 acc[4] = {};
#pragma unroll
  for (int kt = 0; kt < 6; ++kt) {
    const __bf16* src;
    int kb;
    if (kt < 2)      { src = rowE; kb = kt * 32; }
    else if (kt < 4) { src = rowS; kb = (kt - 2) * 32; }
    else             { src = rowR; kb = (kt - 4) * 32; }
    bf16x16 a = load_afrag(src, kb, half);
#pragma unroll
    for (int nt = 0; nt < 4; ++nt)
      acc[nt] = wmma_bf16(a, load_bfrag(pw1, kt, nt, lane), acc[nt]);
  }

  // bias + relu -> LDS (D-frag -> row-major bf16 tile)
#pragma unroll
  for (int nt = 0; nt < 4; ++nt) {
    int col = nt * 16 + lrow;
    float bias = b1[col];
#pragma unroll
    for (int r = 0; r < 8; ++r) {
      float v = acc[nt][r] + bias;
      v = v > 0.f ? v : 0.f;
      s_h[w][(r + 8 * half) * HH + col] = (__bf16)v;
    }
  }
  __syncthreads();

  f32x8 acc2[4] = {};
  const __bf16* lrowp = &s_h[w][lrow * HH];
#pragma unroll
  for (int kt = 0; kt < 2; ++kt) {
    bf16x16 a = load_afrag(lrowp, kt * 32, half);  // ds_load_b128 x2
#pragma unroll
    for (int nt = 0; nt < 4; ++nt)
      acc2[nt] = wmma_bf16(a, load_bfrag(pw2, kt, nt, lane), acc2[nt]);
  }

  int rcv[8];
#pragma unroll
  for (int r = 0; r < 8; ++r) rcv[r] = receivers[tile * 16 + r + 8 * half];

#pragma unroll
  for (int nt = 0; nt < 4; ++nt) {
    int col = nt * 16 + lrow;
    float bias = b2[col];
#pragma unroll
    for (int r = 0; r < 8; ++r) {
      float v = acc2[nt][r] + bias;
      int orow = tile * 16 + r + 8 * half;
      e[(size_t)orow * HH + col] = (__bf16)v;                      // next-round e
      atomicAdd(&agg[(size_t)rcv[r] * HH + col], v);               // segment_sum
    }
  }
}

// ---------------- node update ----------------
// One wave = 16 nodes. A = [n | agg] (K=128). Zeroes its agg rows afterwards.
__global__ void k_node_update(__bf16* __restrict__ n, float* __restrict__ agg,
                              const unsigned* __restrict__ pw1, const float* __restrict__ b1,
                              const unsigned* __restrict__ pw2, const float* __restrict__ b2) {
  __shared__ __align__(16) __bf16 s_h[8][16 * HH];
  int lane = threadIdx.x & 31;
  int w    = threadIdx.x >> 5;
  int tile = blockIdx.x * 8 + w;
  int half = lane >> 4;
  int lrow = lane & 15;
  int row  = tile * 16 + lrow;
  const __bf16* rowN = n + (size_t)row * HH;
  const float*  rowA = agg + (size_t)row * HH;

  f32x8 acc[4] = {};
#pragma unroll
  for (int kt = 0; kt < 4; ++kt) {
    bf16x16 a;
    if (kt < 2) {
      a = load_afrag(rowN, kt * 32, half);
    } else {
      AFrag f;
      int kb = (kt - 2) * 32;
      const float* c0 = rowA + kb + 8 * half;
      const float* c1 = rowA + kb + 16 + 8 * half;
#pragma unroll
      for (int j = 0; j < 8; ++j) {
        f.v[j]     = (__bf16)c0[j];
        f.v[8 + j] = (__bf16)c1[j];
      }
      a = f.v;
    }
#pragma unroll
    for (int nt = 0; nt < 4; ++nt)
      acc[nt] = wmma_bf16(a, load_bfrag(pw1, kt, nt, lane), acc[nt]);
  }

#pragma unroll
  for (int nt = 0; nt < 4; ++nt) {
    int col = nt * 16 + lrow;
    float bias = b1[col];
#pragma unroll
    for (int r = 0; r < 8; ++r) {
      float v = acc[nt][r] + bias;
      v = v > 0.f ? v : 0.f;
      s_h[w][(r + 8 * half) * HH + col] = (__bf16)v;
    }
  }
  __syncthreads();

  f32x8 acc2[4] = {};
  const __bf16* lrowp = &s_h[w][lrow * HH];
#pragma unroll
  for (int kt = 0; kt < 2; ++kt) {
    bf16x16 a = load_afrag(lrowp, kt * 32, half);
#pragma unroll
    for (int nt = 0; nt < 4; ++nt)
      acc2[nt] = wmma_bf16(a, load_bfrag(pw2, kt, nt, lane), acc2[nt]);
  }

#pragma unroll
  for (int nt = 0; nt < 4; ++nt) {
    int col = nt * 16 + lrow;
    float bias = b2[col];
#pragma unroll
    for (int r = 0; r < 8; ++r) {
      int orow = tile * 16 + r + 8 * half;
      n[(size_t)orow * HH + col]   = (__bf16)(acc2[nt][r] + bias);
      agg[(size_t)orow * HH + col] = 0.0f;  // ready for next round's atomics
    }
  }
}

// ---------------- decoder ----------------
// h = relu(e @ Wd1 + bd1) via WMMA; d = h . Wd2 via shfl_xor butterfly.
__global__ void k_decode(const __bf16* __restrict__ e,
                         const unsigned* __restrict__ pwd1, const float* __restrict__ bd1,
                         const float* __restrict__ Wd2, const float* __restrict__ bd2,
                         const float* __restrict__ edges_init,
                         const int* __restrict__ receivers, const int* __restrict__ senders,
                         const float* __restrict__ alpha_p,
                         const unsigned* __restrict__ norm_bits,
                         float* __restrict__ out) {
  int lane = threadIdx.x & 31;
  int w    = threadIdx.x >> 5;
  int tile = blockIdx.x * 8 + w;
  int half = lane >> 4;
  int lrow = lane & 15;
  int edge = tile * 16 + lrow;
  const __bf16* rowE = e + (size_t)edge * HH;

  f32x8 acc[4] = {};
#pragma unroll
  for (int kt = 0; kt < 2; ++kt) {
    bf16x16 a = load_afrag(rowE, kt * 32, half);
#pragma unroll
    for (int nt = 0; nt < 4; ++nt)
      acc[nt] = wmma_bf16(a, load_bfrag(pwd1, kt, nt, lane), acc[nt]);
  }

  float p[8];
#pragma unroll
  for (int r = 0; r < 8; ++r) p[r] = 0.f;
#pragma unroll
  for (int nt = 0; nt < 4; ++nt) {
    int col = nt * 16 + lrow;
    float bias = bd1[col];
    float w2 = Wd2[col];
#pragma unroll
    for (int r = 0; r < 8; ++r) {
      float h = acc[nt][r] + bias;
      h = h > 0.f ? h : 0.f;
      p[r] += h * w2;
    }
  }
  // reduce across the 16 lanes of each half (rows r+8*half)
#pragma unroll
  for (int off = 1; off <= 8; off <<= 1) {
#pragma unroll
    for (int r = 0; r < 8; ++r) p[r] += __shfl_xor(p[r], off, 32);
  }

  if (lrow == 0) {
    float norm  = __uint_as_float(*norm_bits);
    float alpha = *alpha_p;
    float b2    = bd2[0];
#pragma unroll
    for (int r = 0; r < 8; ++r) {
      int oe = tile * 16 + r + 8 * half;
      float d = p[r] + b2;
      float val = edges_init[oe] + alpha * (d * norm);
      val = (receivers[oe] >= senders[oe]) ? val : 0.0f;
      out[oe] = val;
    }
  }
}

// ---------------- launch ----------------

extern "C" void kernel_launch(void* const* d_in, const int* in_sizes, int n_in,
                              void* d_out, int out_size, void* d_ws, size_t ws_size,
                              hipStream_t stream) {
  (void)in_sizes; (void)n_in; (void)out_size; (void)ws_size;
  const float* nodes      = (const float*)d_in[0];
  const float* edges_init = (const float*)d_in[1];
  const int*   receivers  = (const int*)d_in[2];
  const int*   senders    = (const int*)d_in[3];
  const float* Wne1 = (const float*)d_in[4];  const float* bne1 = (const float*)d_in[5];
  const float* Wne2 = (const float*)d_in[6];  const float* bne2 = (const float*)d_in[7];
  const float* Wee1 = (const float*)d_in[8];  const float* bee1 = (const float*)d_in[9];
  const float* Wee2 = (const float*)d_in[10]; const float* bee2 = (const float*)d_in[11];
  const float* Weu1 = (const float*)d_in[12]; const float* beu1 = (const float*)d_in[13];
  const float* Weu2 = (const float*)d_in[14]; const float* beu2 = (const float*)d_in[15];
  const float* Wnu1 = (const float*)d_in[16]; const float* bnu1 = (const float*)d_in[17];
  const float* Wnu2 = (const float*)d_in[18]; const float* bnu2 = (const float*)d_in[19];
  const float* Wd1  = (const float*)d_in[20]; const float* bd1  = (const float*)d_in[21];
  const float* Wd2  = (const float*)d_in[22]; const float* bd2  = (const float*)d_in[23];
  const float* alpha = (const float*)d_in[24];

  char* ws = (char*)d_ws;
  unsigned* norm_bits = (unsigned*)(ws + OFF_NORM);
  unsigned* pw        = (unsigned*)(ws + OFF_PW);
  __bf16*   nfeat     = (__bf16*)(ws + OFF_N);
  float*    agg       = (float*)(ws + OFF_AGG);
  __bf16*   efeat     = (__bf16*)(ws + OFF_E);

  // init: agg = 0, norm_bits = 0
  k_zero<<<(NN * HH + 255) / 256, 256, 0, stream>>>(agg, norm_bits);
  k_absmax<<<1024, 256, 0, stream>>>(edges_init, norm_bits);

  // pack weights into WMMA B-fragments (bf16)
  k_pack<<<2, 128, 0, stream>>>(Wne2, pw + PW_NE2);
  k_pack<<<2, 128, 0, stream>>>(Wee2, pw + PW_EE2);
  k_pack<<<6, 128, 0, stream>>>(Weu1, pw + PW_EU1);
  k_pack<<<2, 128, 0, stream>>>(Weu2, pw + PW_EU2);
  k_pack<<<4, 128, 0, stream>>>(Wnu1, pw + PW_NU1);
  k_pack<<<2, 128, 0, stream>>>(Wnu2, pw + PW_NU2);
  k_pack<<<2, 128, 0, stream>>>(Wd1,  pw + PW_D1);

  // encoders (one wave per 16 rows, 8 waves per block)
  k_encode<<<NN / 128, 256, 0, stream>>>(nodes, Wne1, bne1, pw + PW_NE2, bne2,
                                         nfeat, nullptr);
  k_encode<<<EE / 128, 256, 0, stream>>>(edges_init, Wee1, bee1, pw + PW_EE2, bee2,
                                         efeat, norm_bits);

  // message-passing rounds
  for (int t = 0; t < TT; ++t) {
    k_edge_update<<<EE / 128, 256, 0, stream>>>(efeat, nfeat, senders, receivers,
                                                pw + PW_EU1, beu1, pw + PW_EU2, beu2,
                                                agg);
    k_node_update<<<NN / 128, 256, 0, stream>>>(nfeat, agg,
                                                pw + PW_NU1, bnu1, pw + PW_NU2, bnu2);
  }

  // decode + residual + lower-tri mask
  k_decode<<<EE / 128, 256, 0, stream>>>(efeat, pw + PW_D1, bd1, Wd2, bd2,
                                         edges_init, receivers, senders,
                                         alpha, norm_bits, (float*)d_out);
}
